// TFMultiHeadAttention_53386443489959
// MI455X (gfx1250) — matrix operand order repaired and checked by reading, not compiled, not run on hardware
//
#include <hip/hip_runtime.h>
#include <stdint.h>

#define BB 2
#define SS 2048
#define DD 1280
#define HH 16
#define DEPTH 80
#define DP 96            // Q/K depth padded to 3 * 32 (zeros in pad)
#define MTOT (BB * SS)   // 4096

typedef __attribute__((ext_vector_type(16))) __bf16          v16bf;
typedef __attribute__((ext_vector_type(8)))  float           v8f;
typedef __attribute__((ext_vector_type(16))) unsigned short  v16u;
typedef __attribute__((ext_vector_type(8)))  unsigned short  v8u;

union Frag {
    v8u   h[2];
    v16u  u;
    v16bf b;
};

__device__ __forceinline__ unsigned short f2bf(float f) {
    union { float f; unsigned u; } c;
    c.f = f;
    unsigned u = c.u;
    unsigned r = u + 0x7fffu + ((u >> 16) & 1u);   // round-to-nearest-even
    return (unsigned short)(r >> 16);
}

__device__ __forceinline__ unsigned ldsoff(const void* p) {
    return (unsigned)(uintptr_t)p;
}

// CDNA5 async global->LDS copy (16 bytes), tracked by ASYNCcnt.
__device__ __forceinline__ void cp_async16(unsigned lds_off, const void* g) {
    asm volatile("global_load_async_to_lds_b128 %0, %1, off"
                 :: "v"(lds_off), "v"((unsigned long long)(uintptr_t)g)
                 : "memory");
}
__device__ __forceinline__ void cp_async_wait() {
    asm volatile("s_wait_asynccnt 0x0" ::: "memory");
}

__device__ __forceinline__ v8f wmma_bf16(const Frag& a, const Frag& b, v8f c) {
    return __builtin_amdgcn_wmma_f32_16x16x32_bf16(
        /*neg_a=*/false, a.b, /*neg_b=*/false, b.b,
        /*c_mod=*/(short)0, c, /*reuse_a=*/false, /*reuse_b=*/false);
}

// ---------------------------------------------------------------- converts
__global__ void k_cvt_bf16(const float* __restrict__ src,
                           unsigned short* __restrict__ dst, int n) {
    int i = blockIdx.x * 256 + threadIdx.x;
    if (i < n) dst[i] = f2bf(src[i]);
}

// convert weight [k][n] f32 -> transposed bf16 [n][k] (so GEMM B tiles are
// contiguous in k and can be staged purely with async-to-LDS)
__global__ void k_cvt_bf16_T(const float* __restrict__ src,
                             unsigned short* __restrict__ dst) {
    int i = blockIdx.x * 256 + threadIdx.x;   // DD*DD exactly
    int r = i / DD, c = i % DD;
    dst[(size_t)c * DD + r] = f2bf(src[i]);
}

// zero the padded depth columns (d = 80..95) of Q and K
__global__ void k_zero_pad(unsigned short* __restrict__ Q,
                           unsigned short* __restrict__ K) {
    int i = blockIdx.x * 256 + threadIdx.x;   // over BB*HH*SS*16
    int s = i >> 4, d = DEPTH + (i & 15);
    Q[(size_t)s * DP + d] = 0;
    K[(size_t)s * DP + d] = 0;
}

// ---------------------------------------------------------------- GEMM
// C[M=4096][N=1280] = A(bf16,[m][k]) @ Wt(bf16,[n][k])^T + bias, epilogue:
//  mode 0: Q  -> [B,H,S,DP] bf16, scaled by 1/sqrt(80)
//  mode 1: K  -> [B,H,S,DP] bf16
//  mode 2: V  -> [B,H,DEPTH,S] bf16 (transposed for P@V)
//  mode 3: out-> [M,N] f32
__global__ __launch_bounds__(256)
void k_gemm(const unsigned short* __restrict__ A,
            const unsigned short* __restrict__ Wt,
            const float* __restrict__ bias,
            void* __restrict__ out, int mode, float scale) {
    __shared__ unsigned short At[2][128 * 32];   // [m][k] 2 x 8 KB
    __shared__ unsigned short Bt[2][64 * 32];    // [n][k] 2 x 4 KB

    const int tid  = threadIdx.x;
    const int lane = tid & 31, wid = tid >> 5;
    const int ml   = lane & 15, half = lane >> 4;
    const int wm   = wid >> 1,  wn   = wid & 1;
    const int m0   = blockIdx.x * 128, n0 = blockIdx.y * 64;

    auto stage = [&](int buf, int kc) {
        // A tile: 128x32 bf16 = 512 x 16B async chunks (2 per thread)
        for (int c = tid; c < 512; c += 256) {
            int row = c >> 2, col = c & 3;
            cp_async16(ldsoff(&At[buf][row * 32 + col * 8]),
                       &A[(size_t)(m0 + row) * DD + kc * 32 + col * 8]);
        }
        // B tile: 64x32 bf16 = 256 x 16B async chunks (1 per thread)
        {
            int n = tid >> 2, col = tid & 3;
            cp_async16(ldsoff(&Bt[buf][n * 32 + col * 8]),
                       &Wt[(size_t)(n0 + n) * DD + kc * 32 + col * 8]);
        }
    };

    v8f acc[2][2] = {};

    stage(0, 0);
    cp_async_wait();
    __syncthreads();

    const int KT = DD / 32;   // 40
    for (int kc = 0; kc < KT; ++kc) {
        const int cur = kc & 1;
        if (kc + 1 < KT) stage(1 - cur, kc + 1);   // overlaps with WMMA below

        Frag af[2], bf[2];
        for (int i = 0; i < 2; ++i) {
            const unsigned short* rp = &At[cur][(wm * 32 + i * 16 + ml) * 32];
            af[i].h[0] = *(const v8u*)(rp + half * 8);
            af[i].h[1] = *(const v8u*)(rp + 16 + half * 8);
        }
        for (int j = 0; j < 2; ++j) {
            const unsigned short* rp = &Bt[cur][(wn * 32 + j * 16 + ml) * 32];
            bf[j].h[0] = *(const v8u*)(rp + half * 8);
            bf[j].h[1] = *(const v8u*)(rp + 16 + half * 8);
        }
        for (int i = 0; i < 2; ++i)
            for (int j = 0; j < 2; ++j)
                acc[i][j] = wmma_bf16(af[i], bf[j], acc[i][j]);

        cp_async_wait();
        __syncthreads();
    }

    for (int i = 0; i < 2; ++i)
        for (int j = 0; j < 2; ++j) {
            int n = n0 + wn * 32 + j * 16 + ml;
            float bv = bias[n];
            for (int r = 0; r < 8; ++r) {
                int m = m0 + wm * 32 + i * 16 + r + 8 * half;
                float v = (acc[i][j][r] + bv) * scale;
                if (mode == 3) {
                    ((float*)out)[(size_t)m * DD + n] = v;
                } else {
                    int b = m / SS, s = m % SS;
                    int h = n / DEPTH, d = n % DEPTH;
                    unsigned short* o = (unsigned short*)out;
                    if (mode == 2)
                        o[((size_t)(b * HH + h) * DEPTH + d) * SS + s] = f2bf(v);
                    else
                        o[((size_t)(b * HH + h) * SS + s) * DP + d] = f2bf(v);
                }
            }
        }
}

// ---------------------------------------------------------------- attention
// Flash attention, causal (reference mask == strict-upper * -10000).
// Grid: (S/128, B*H); 8 waves, each owns 16 query rows.
__global__ __launch_bounds__(256)
void k_attn(const unsigned short* __restrict__ Q,   // [BH][S][DP]
            const unsigned short* __restrict__ K,   // [BH][S][DP]
            const unsigned short* __restrict__ Vt,  // [BH][DEPTH][S]
            unsigned short* __restrict__ ctx) {     // [B][S][D]
    __shared__ unsigned short Kt[2][32 * DP];      // [key][d]  2 x 6 KB
    __shared__ unsigned short Vtt[2][DEPTH * 32];  // [d][key]  2 x 5 KB
    __shared__ unsigned short Pb[8 * 16 * 32];     // per-wave P, 8 KB

    const int tid  = threadIdx.x, lane = tid & 31, wid = tid >> 5;
    const int ml   = lane & 15,  half = lane >> 4;
    const int q0   = blockIdx.x * 128;
    const int bh   = blockIdx.y;
    const int qrow = q0 + wid * 16;

    auto stage = [&](int buf, int kt) {
        // K tile: 32 x 96 bf16 = 384 x 16B chunks
        for (int c = tid; c < 384; c += 256) {
            int row = c / 12, col = c % 12;
            cp_async16(ldsoff(&Kt[buf][row * DP + col * 8]),
                       &K[((size_t)bh * SS + kt * 32 + row) * DP + col * 8]);
        }
        // V tile (pre-transposed): 80 x 32 bf16 = 320 chunks
        for (int c = tid; c < 320; c += 256) {
            int row = c >> 1, col = c & 1;
            cp_async16(ldsoff(&Vtt[buf][row * 32 + col * 8]),
                       &Vt[((size_t)bh * DEPTH + row) * SS + kt * 32 + col * 8]);
        }
    };

    // Q fragments stay in registers for the whole K loop
    Frag qf[3];
    {
        const unsigned short* rp = &Q[((size_t)bh * SS + qrow + ml) * DP];
        for (int c = 0; c < 3; ++c) {
            qf[c].h[0] = *(const v8u*)(rp + c * 32 + half * 8);
            qf[c].h[1] = *(const v8u*)(rp + c * 32 + 16 + half * 8);
        }
    }

    float mrow[8], lrow[8];
    v8f O[5] = {};
    for (int r = 0; r < 8; ++r) { mrow[r] = -3.0e38f; lrow[r] = 0.f; }

    const int ktiles = q0 / 32 + 4;   // causal: keys <= q0+127
    stage(0, 0);
    cp_async_wait();
    __syncthreads();

    for (int kt = 0; kt < ktiles; ++kt) {
        const int cur = kt & 1;
        if (kt + 1 < ktiles) stage(1 - cur, kt + 1);   // overlap with compute

        // scores S = Q @ K^T : 16 x 32 as two 16x16 n-tiles, 3 K-chunks each
        v8f s[2] = {};
        for (int nt = 0; nt < 2; ++nt)
            for (int c = 0; c < 3; ++c) {
                Frag kf;
                kf.h[0] = *(const v8u*)&Kt[cur][(nt * 16 + ml) * DP + c * 32 + half * 8];
                kf.h[1] = *(const v8u*)&Kt[cur][(nt * 16 + ml) * DP + c * 32 + 16 + half * 8];
                s[nt] = wmma_bf16(qf[c], kf, s[nt]);
            }

        // causal mask (reference adds -10000, not -inf)
        for (int nt = 0; nt < 2; ++nt) {
            int kcol = kt * 32 + nt * 16 + ml;
            for (int r = 0; r < 8; ++r)
                if (kcol > qrow + r + 8 * half) s[nt][r] += -10000.0f;
        }

        // online softmax per row (rows r+8*half live across a 16-lane half)
        for (int r = 0; r < 8; ++r) {
            float v = fmaxf(s[0][r], s[1][r]);
            for (int m = 1; m < 16; m <<= 1) v = fmaxf(v, __shfl_xor(v, m, 32));
            float mnew = fmaxf(mrow[r], v);
            float corr = __expf(mrow[r] - mnew);
            float p0 = __expf(s[0][r] - mnew);
            float p1 = __expf(s[1][r] - mnew);
            float rs = p0 + p1;
            for (int m = 1; m < 16; m <<= 1) rs += __shfl_xor(rs, m, 32);
            lrow[r] = lrow[r] * corr + rs;
            mrow[r] = mnew;
            for (int nt = 0; nt < 5; ++nt) O[nt][r] *= corr;
            unsigned short* pp = &Pb[(wid * 16 + r + 8 * half) * 32];
            pp[0 * 16 + ml] = f2bf(p0);
            pp[1 * 16 + ml] = f2bf(p1);
        }

        // re-fragment P as WMMA A operand (per-wave LDS, DScnt-ordered)
        Frag pf;
        {
            const unsigned short* rp = &Pb[(wid * 16 + ml) * 32];
            pf.h[0] = *(const v8u*)(rp + half * 8);
            pf.h[1] = *(const v8u*)(rp + 16 + half * 8);
        }
        for (int nt = 0; nt < 5; ++nt) {
            Frag vf;
            vf.h[0] = *(const v8u*)&Vtt[cur][(nt * 16 + ml) * 32 + half * 8];
            vf.h[1] = *(const v8u*)&Vtt[cur][(nt * 16 + ml) * 32 + 16 + half * 8];
            O[nt] = wmma_bf16(pf, vf, O[nt]);
        }

        cp_async_wait();
        __syncthreads();
    }

    // normalize + store ctx (merged heads, bf16)
    const int b = bh / HH, h = bh % HH;
    for (int nt = 0; nt < 5; ++nt)
        for (int r = 0; r < 8; ++r) {
            int srow = qrow + r + 8 * half;
            float o = O[nt][r] / lrow[r];
            ctx[(size_t)(b * SS + srow) * DD + h * DEPTH + nt * 16 + ml] = f2bf(o);
        }
}

// ---------------------------------------------------------------- launcher
extern "C" void kernel_launch(void* const* d_in, const int* in_sizes, int n_in,
                              void* d_out, int out_size, void* d_ws, size_t ws_size,
                              hipStream_t stream) {
    const float* x  = (const float*)d_in[0];
    // d_in[1] = mask: causal, reproduced analytically in-kernel
    const float* Wq = (const float*)d_in[2];
    const float* bq = (const float*)d_in[3];
    const float* Wk = (const float*)d_in[4];
    const float* bk = (const float*)d_in[5];
    const float* Wv = (const float*)d_in[6];
    const float* bv = (const float*)d_in[7];
    const float* Wo = (const float*)d_in[8];
    const float* bo = (const float*)d_in[9];
    float* out = (float*)d_out;

    char* ws = (char*)d_ws;
    size_t off = 0;
    auto alloc = [&](size_t bytes) -> char* {
        char* p = ws + off;
        off = (off + bytes + 255) & ~(size_t)255;
        return p;
    };
    unsigned short* xb   = (unsigned short*)alloc((size_t)MTOT * DD * 2);
    unsigned short* Wqt  = (unsigned short*)alloc((size_t)DD * DD * 2);
    unsigned short* Wkt  = (unsigned short*)alloc((size_t)DD * DD * 2);
    unsigned short* Wvt  = (unsigned short*)alloc((size_t)DD * DD * 2);
    unsigned short* Wot  = (unsigned short*)alloc((size_t)DD * DD * 2);
    unsigned short* Qb   = (unsigned short*)alloc((size_t)BB * HH * SS * DP * 2);
    unsigned short* Kb   = (unsigned short*)alloc((size_t)BB * HH * SS * DP * 2);
    unsigned short* Vtb  = (unsigned short*)alloc((size_t)BB * HH * DEPTH * SS * 2);
    unsigned short* ctxb = (unsigned short*)alloc((size_t)MTOT * DD * 2);

    k_cvt_bf16<<<(MTOT * DD + 255) / 256, 256, 0, stream>>>(x, xb, MTOT * DD);
    k_cvt_bf16_T<<<(DD * DD) / 256, 256, 0, stream>>>(Wq, Wqt);
    k_cvt_bf16_T<<<(DD * DD) / 256, 256, 0, stream>>>(Wk, Wkt);
    k_cvt_bf16_T<<<(DD * DD) / 256, 256, 0, stream>>>(Wv, Wvt);
    k_cvt_bf16_T<<<(DD * DD) / 256, 256, 0, stream>>>(Wo, Wot);
    k_zero_pad<<<(BB * HH * SS * 16) / 256, 256, 0, stream>>>(Qb, Kb);

    dim3 gg(MTOT / 128, DD / 64);
    const float qscale = 0.11180339887498949f;   // 1/sqrt(80)
    k_gemm<<<gg, 256, 0, stream>>>(xb, Wqt, bq, Qb, 0, qscale);
    k_gemm<<<gg, 256, 0, stream>>>(xb, Wkt, bk, Kb, 1, 1.0f);
    k_gemm<<<gg, 256, 0, stream>>>(xb, Wvt, bv, Vtb, 2, 1.0f);

    k_attn<<<dim3(SS / 128, BB * HH), 256, 0, stream>>>(Qb, Kb, Vtb, ctxb);

    k_gemm<<<gg, 256, 0, stream>>>(ctxb, Wot, bo, out, 3, 1.0f);
}
